// CustomModel_13657996001616
// MI455X (gfx1250) — compile-verified, matrix operand order checked by reading, not run on hardware
//
#include <hip/hip_runtime.h>
#include <hip/hip_bf16.h>

typedef __attribute__((ext_vector_type(16))) __bf16 v16bf;
typedef __attribute__((ext_vector_type(8)))  float  v8f;

__device__ __forceinline__ unsigned short f32_to_bf16(float f) {
    unsigned int u = __float_as_uint(f);
    unsigned int r = u + 0x7FFFu + ((u >> 16) & 1u);   // round-to-nearest-even
    return (unsigned short)(r >> 16);
}

// ---------------------------------------------------------------------------
// Tiled bf16 WMMA GEMM: C[M,N] = op(A[M,K] @ B[K,N] + bias), fp32 accumulate.
// Block tile 128(M) x 64(N), K-step 32. 8 waves: 4(M) x 2(N), each wave does a
// 32x32 subtile = 2x2 v_wmma_f32_16x16x32_bf16. K must be a multiple of 32,
// N a multiple of 4. Out-of-range rows/cols are CLAMPED during staging (their
// products land in C entries that the epilogue discards), so the K-loop has no
// divergent guards and all global staging is done with b128 loads.
// ---------------------------------------------------------------------------
#define TILE_M 128
#define TILE_N 64
#define TILE_K 32

__global__ void __launch_bounds__(256)
gemm_bf16_wmma(const float* __restrict__ A, const float* __restrict__ B,
               const float* __restrict__ bias, float* __restrict__ C,
               int M, int N, int K, int relu)
{
    __shared__ unsigned short As[TILE_M][TILE_K];   // [m][k], k contiguous
    __shared__ unsigned short Bs[TILE_N][TILE_K];   // [n][k], k contiguous

    const int tid   = threadIdx.x;
    const int lane  = tid & 31;
    const int wave  = tid >> 5;
    const int waveM = wave & 3;        // 0..3
    const int waveN = wave >> 2;       // 0..1
    const int mBlock = blockIdx.y * TILE_M;
    const int nBlock = blockIdx.x * TILE_N;
    const int l15 = lane & 15;
    const int hi  = lane >> 4;         // 0 or 1
    const int kA  = hi * 8;            // per-half-wave K offset (CDNA5 layout)

    // Clamped staging coordinates (uniform per thread across K-steps)
    const int aM  = (tid >> 3);                 // rows 0..31 handled at i-th pass offset
    const int aK4 = (tid & 7) * 4;
    const int bK  = (tid >> 4);                 // k rows 0..15 per pass
    const int bN4 = (tid & 15) * 4;

    v8f acc[2][2];
    const v8f vzero = {0.f,0.f,0.f,0.f,0.f,0.f,0.f,0.f};
    for (int a = 0; a < 2; ++a)
        for (int b = 0; b < 2; ++b) acc[a][b] = vzero;

    union FragU { unsigned int u[8]; v16bf v; };

    for (int kk = 0; kk < K; kk += TILE_K) {
        // ---- Stage A tile: 128x32 f32 -> bf16, 4 x float4 per thread ----
        #pragma unroll
        for (int i = 0; i < 4; ++i) {
            int m   = i * 32 + aM;
            int row = mBlock + m;
            if (row >= M) row = M - 1;                       // clamp (discarded later)
            const float4 v = *(const float4*)(A + (size_t)row * K + kk + aK4);
            ushort4 p;
            p.x = f32_to_bf16(v.x); p.y = f32_to_bf16(v.y);
            p.z = f32_to_bf16(v.z); p.w = f32_to_bf16(v.w);
            *(ushort4*)&As[m][aK4] = p;
        }
        // ---- Stage B tile transposed: 32x64 f32 -> bf16, 2 x float4 ----
        #pragma unroll
        for (int i = 0; i < 2; ++i) {
            int k   = i * 16 + bK;
            int col = nBlock + bN4;
            if (col + 4 > N) col = N - 4;                    // clamp (N % 4 == 0)
            const float4 v = *(const float4*)(B + (size_t)(kk + k) * N + col);
            Bs[bN4 + 0][k] = f32_to_bf16(v.x);
            Bs[bN4 + 1][k] = f32_to_bf16(v.y);
            Bs[bN4 + 2][k] = f32_to_bf16(v.z);
            Bs[bN4 + 3][k] = f32_to_bf16(v.w);
        }
        // Prefetch next K tile while waves compute this one
        if (kk + TILE_K < K) {
            int row = mBlock + aM; if (row >= M) row = M - 1;
            __builtin_prefetch(A + (size_t)row * K + kk + TILE_K + aK4, 0, 0);
            int col = nBlock + bN4; if (col + 4 > N) col = N - 4;
            __builtin_prefetch(B + (size_t)(kk + TILE_K + bK) * N + col, 0, 0);
        }
        __syncthreads();

        // Build fragments per CDNA5 16-bit layout:
        // lanes 0-15: K = {0..7, 16..23}; lanes 16-31: K = {8..15, 24..31}
        FragU afrag[2], bfrag[2];
        #pragma unroll
        for (int t = 0; t < 2; ++t) {
            const unsigned short* ap = &As[waveM * 32 + t * 16 + l15][0];
            const unsigned short* bp = &Bs[waveN * 32 + t * 16 + l15][0];
            #pragma unroll
            for (int j = 0; j < 4; ++j) {
                afrag[t].u[j]     = *(const unsigned int*)(ap + kA + 2 * j);
                afrag[t].u[4 + j] = *(const unsigned int*)(ap + 16 + kA + 2 * j);
                bfrag[t].u[j]     = *(const unsigned int*)(bp + kA + 2 * j);
                bfrag[t].u[4 + j] = *(const unsigned int*)(bp + 16 + kA + 2 * j);
            }
        }
        #pragma unroll
        for (int tm = 0; tm < 2; ++tm)
            #pragma unroll
            for (int tn = 0; tn < 2; ++tn)
                acc[tm][tn] = __builtin_amdgcn_wmma_f32_16x16x32_bf16(
                    false, afrag[tm].v, false, bfrag[tn].v,
                    (short)0, acc[tm][tn], false, false);
        __syncthreads();
    }

    // Epilogue: C VGPR r -> M = r + 8*hi ; N = lane&15
    for (int tm = 0; tm < 2; ++tm) {
        for (int tn = 0; tn < 2; ++tn) {
            int col = nBlock + waveN * 32 + tn * 16 + l15;
            if (col >= N) continue;
            float bv = bias ? bias[col] : 0.0f;
            #pragma unroll
            for (int r = 0; r < 8; ++r) {
                int row = mBlock + waveM * 32 + tm * 16 + hi * 8 + r;
                if (row < M) {
                    float v = acc[tm][tn][r] + bv;
                    if (relu) v = fmaxf(v, 0.0f);
                    C[(size_t)row * N + col] = v;
                }
            }
        }
    }
}

// ---------------------------------------------------------------------------
// Helper kernels
// ---------------------------------------------------------------------------
__global__ void zero_kernel(float* __restrict__ p, size_t n) {
    size_t i = (size_t)blockIdx.x * blockDim.x + threadIdx.x;
    if (i < n) p[i] = 0.0f;
}

__global__ void deg_kernel(const long long* __restrict__ dst,
                           float* __restrict__ deg, long long E) {
    long long e = (long long)blockIdx.x * blockDim.x + threadIdx.x;
    if (e < E) unsafeAtomicAdd(&deg[dst[e]], 1.0f);
}

__global__ void dinv_kernel(const float* __restrict__ deg,
                            float* __restrict__ dinv, int n) {
    int i = blockIdx.x * blockDim.x + threadIdx.x;
    if (i < n) {
        float d = deg[i];
        dinv[i] = (d > 0.0f) ? rsqrtf(d) : 0.0f;
    }
}

// out[i] = sum_j W[i*cols + j]   (W1.sum(-1) / W2.sum(-1))
__global__ void wsum_kernel(const float* __restrict__ W, float* __restrict__ out,
                            int rows, int cols) {
    int i = blockIdx.x * blockDim.x + threadIdx.x;
    if (i < rows) {
        float s = 0.0f;
        for (int j = 0; j < cols; ++j) s += W[(size_t)i * cols + j];
        out[i] = s;
    }
}

// Router: logits = H2 @ w3 + b3 ; hard argmax -> softened one-hot m[N,3]
__global__ void __launch_bounds__(256)
router_kernel(const float* __restrict__ H2, const float* __restrict__ w3,
              const float* __restrict__ b3, float* __restrict__ m_out, int Nn) {
    int node = blockIdx.x * 8 + (threadIdx.x >> 5);
    if (node >= Nn) return;
    int lane = threadIdx.x & 31;
    float a0 = 0.f, a1 = 0.f, a2 = 0.f;
    const float* hrow = H2 + (size_t)node * 512;
    for (int j = lane; j < 512; j += 32) {
        float h = hrow[j];
        a0 += h * w3[j * 3 + 0];
        a1 += h * w3[j * 3 + 1];
        a2 += h * w3[j * 3 + 2];
    }
    for (int off = 16; off > 0; off >>= 1) {
        a0 += __shfl_xor(a0, off);
        a1 += __shfl_xor(a1, off);
        a2 += __shfl_xor(a2, off);
    }
    if (lane == 0) {
        float l0 = a0 + b3[0], l1 = a1 + b3[1], l2 = a2 + b3[2];
        int arg = 0; float best = l0;
        if (l1 > best) { best = l1; arg = 1; }
        if (l2 > best) { best = l2; arg = 2; }
        const float hiw = 0.7f, low = (1.0f - 0.7f) * 0.5f;
        m_out[(size_t)node * 3 + 0] = (arg == 0) ? hiw : low;
        m_out[(size_t)node * 3 + 1] = (arg == 1) ? hiw : low;
        m_out[(size_t)node * 3 + 2] = (arg == 2) ? hiw : low;
    }
}

// result1 = (m @ W1s) * x2 + m @ bp1     [N,768]
__global__ void result1_kernel(const float* __restrict__ m, const float* __restrict__ W1s,
                               const float* __restrict__ bp1, const float* __restrict__ x2,
                               float* __restrict__ r1, int Nn) {
    size_t idx = (size_t)blockIdx.x * blockDim.x + threadIdx.x;
    size_t total = (size_t)Nn * 768;
    if (idx >= total) return;
    size_t i = idx / 768; int f = (int)(idx % 768);
    float m0 = m[i * 3], m1 = m[i * 3 + 1], m2 = m[i * 3 + 2];
    float a = m0 * W1s[f] + m1 * W1s[768 + f] + m2 * W1s[1536 + f];
    float b = m0 * bp1[f] + m1 * bp1[768 + f] + m2 * bp1[1536 + f];
    r1[idx] = a * x2[idx] + b;
}

// Edge scatter: agg[dst] += dinv[src]*dinv[dst] * h[src].
// F = 4 << shift features; (256 >> shift) edges per 256-thread block.
__global__ void __launch_bounds__(256)
scatter_kernel(const float* __restrict__ h, const long long* __restrict__ src,
               const long long* __restrict__ dst, const float* __restrict__ dinv,
               float* __restrict__ agg, long long E, int shift) {
    int tid = threadIdx.x;
    long long e = (long long)blockIdx.x * (256 >> shift) + (tid >> shift);
    if (e >= E) return;
    int f = (tid & ((1 << shift) - 1)) * 4;
    int F = 4 << shift;
    long long s = src[e], d = dst[e];
    float nrm = dinv[s] * dinv[d];
    const float4 v = *(const float4*)(h + (size_t)s * F + f);
    float* ap = agg + (size_t)d * F + f;
    unsafeAtomicAdd(ap + 0, v.x * nrm);
    unsafeAtomicAdd(ap + 1, v.y * nrm);
    unsafeAtomicAdd(ap + 2, v.z * nrm);
    unsafeAtomicAdd(ap + 3, v.w * nrm);
}

// g1 = (m @ W12) * relu(agg1 + gcn1_b) + 2e-4 * (result1 @ W13)  (in-place on agg1)
__global__ void g1_kernel(const float* __restrict__ m, const float* __restrict__ W12,
                          const float* __restrict__ gcn1_b, const float* __restrict__ r1w13,
                          float* __restrict__ agg_g1, int Nn) {
    size_t idx = (size_t)blockIdx.x * blockDim.x + threadIdx.x;
    size_t total = (size_t)Nn * 256;
    if (idx >= total) return;
    size_t i = idx / 256; int f = (int)(idx % 256);
    float s = m[i * 3] * W12[f] + m[i * 3 + 1] * W12[256 + f] + m[i * 3 + 2] * W12[512 + f];
    float g = fmaxf(agg_g1[idx] + gcn1_b[f], 0.0f);
    agg_g1[idx] = s * g + 2e-4f * r1w13[idx];
}

// result2 = (m @ W2s) * g1 + m @ bp2   [N,256]
__global__ void result2_kernel(const float* __restrict__ m, const float* __restrict__ W2s,
                               const float* __restrict__ bp2, const float* __restrict__ g1,
                               float* __restrict__ r2, int Nn) {
    size_t idx = (size_t)blockIdx.x * blockDim.x + threadIdx.x;
    size_t total = (size_t)Nn * 256;
    if (idx >= total) return;
    size_t i = idx / 256; int f = (int)(idx % 256);
    float m0 = m[i * 3], m1 = m[i * 3 + 1], m2 = m[i * 3 + 2];
    float a = m0 * W2s[f] + m1 * W2s[256 + f] + m2 * W2s[512 + f];
    float b = m0 * bp2[f] + m1 * bp2[256 + f] + m2 * bp2[512 + f];
    r2[idx] = a * g1[idx] + b;
}

// Final: g2 = relu(agg2 + gcn2_b); out = log_softmax(g2 @ fc_w + fc_b). Wave/node.
__global__ void __launch_bounds__(256)
final_kernel(const float* __restrict__ agg2, const float* __restrict__ b2,
             const float* __restrict__ fcw, const float* __restrict__ fcb,
             float* __restrict__ out, int Nn) {
    int node = blockIdx.x * 8 + (threadIdx.x >> 5);
    if (node >= Nn) return;
    int lane = threadIdx.x & 31;
    float g = fmaxf(agg2[(size_t)node * 32 + lane] + b2[lane], 0.0f);
    float a0 = 0.0f, a1 = 0.0f;
    for (int f = 0; f < 32; ++f) {
        float gf = __shfl(g, f);
        a0 += gf * fcw[f * 40 + lane];
        if (lane < 8) a1 += gf * fcw[f * 40 + 32 + lane];
    }
    a0 += fcb[lane];
    float o1 = (lane < 8) ? (a1 + fcb[32 + lane]) : -3.4e38f;
    float mx = fmaxf(a0, o1);
    for (int off = 16; off > 0; off >>= 1) mx = fmaxf(mx, __shfl_xor(mx, off));
    float s = expf(a0 - mx) + ((lane < 8) ? expf(o1 - mx) : 0.0f);
    for (int off = 16; off > 0; off >>= 1) s += __shfl_xor(s, off);
    float lse = mx + logf(s);
    out[(size_t)node * 40 + lane] = a0 - lse;
    if (lane < 8) out[(size_t)node * 40 + 32 + lane] = o1 - lse;
}

// ---------------------------------------------------------------------------
extern "C" void kernel_launch(void* const* d_in, const int* in_sizes, int n_in,
                              void* d_out, int out_size, void* d_ws, size_t ws_size,
                              hipStream_t stream) {
    (void)n_in; (void)out_size; (void)ws_size;
    const float*     x1     = (const float*)d_in[0];
    const float*     x2     = (const float*)d_in[1];
    /* x3 (d_in[2]) is unused by the reference */
    const long long* ei     = (const long long*)d_in[3];
    const float*     mlp_w1 = (const float*)d_in[4];
    const float*     mlp_b1 = (const float*)d_in[5];
    const float*     mlp_w2 = (const float*)d_in[6];
    const float*     mlp_b2 = (const float*)d_in[7];
    const float*     mlp_w3 = (const float*)d_in[8];
    const float*     mlp_b3 = (const float*)d_in[9];
    const float*     W1     = (const float*)d_in[10];
    const float*     W12    = (const float*)d_in[11];
    const float*     W13    = (const float*)d_in[12];
    const float*     bp1    = (const float*)d_in[13];
    const float*     W2     = (const float*)d_in[14];
    const float*     bp2    = (const float*)d_in[15];
    const float*     gcn1_w = (const float*)d_in[16];
    const float*     gcn1_b = (const float*)d_in[17];
    const float*     gcn2_w = (const float*)d_in[18];
    const float*     gcn2_b = (const float*)d_in[19];
    const float*     fc_w   = (const float*)d_in[20];
    const float*     fc_b   = (const float*)d_in[21];

    const int       Nn  = in_sizes[0] / 768;
    const long long E   = (long long)in_sizes[3] / 2;
    const long long* src = ei;
    const long long* dst = ei + E;

    // Workspace layout (floats), with aliasing across pipeline phases:
    float* ws = (float*)d_ws;
    size_t off = 0;
    float* P0 = ws + off; off += (size_t)Nn * 768;   // H1, then result1
    float* P1 = ws + off; off += (size_t)Nn * 512;   // H2, then [h1 | agg1/g1], then [h2 | agg2]
    float* P2 = ws + off; off += (size_t)Nn * 256;   // r1w13, then result2
    float* m    = ws + off; off += (size_t)Nn * 3;
    float* deg  = ws + off; off += (size_t)Nn;
    float* dinv = ws + off; off += (size_t)Nn;
    float* W1s  = ws + off; off += 3 * 768;
    float* W2s  = ws + off; off += 3 * 256;

    float* H1      = P0;
    float* result1 = P0;
    float* H2      = P1;
    float* h1      = P1;
    float* agg1    = P1 + (size_t)Nn * 256;   // also g1 (in-place)
    float* r1w13   = P2;
    float* g1      = agg1;
    float* result2 = P2;
    float* h2      = P1;                      // h1 dead by then
    float* agg2    = P1 + (size_t)Nn * 32;

    auto nb = [](size_t n) { return (unsigned)((n + 255) / 256); };
    dim3 blk(256);

    // Graph normalization
    zero_kernel<<<nb((size_t)Nn), blk, 0, stream>>>(deg, (size_t)Nn);
    deg_kernel<<<nb((size_t)E), blk, 0, stream>>>(dst, deg, E);
    dinv_kernel<<<nb((size_t)Nn), blk, 0, stream>>>(deg, dinv, Nn);

    // Expert-weight reductions W1.sum(-1), W2.sum(-1)
    wsum_kernel<<<nb(3 * 768), blk, 0, stream>>>(W1, W1s, 3 * 768, 256);
    wsum_kernel<<<nb(3 * 256), blk, 0, stream>>>(W2, W2s, 3 * 256, 256);

    // MLP router
    dim3 gmlp((512 + TILE_N - 1) / TILE_N, (Nn + TILE_M - 1) / TILE_M);
    gemm_bf16_wmma<<<gmlp, blk, 0, stream>>>(x1, mlp_w1, mlp_b1, H1, Nn, 512, 768, 1);
    gemm_bf16_wmma<<<gmlp, blk, 0, stream>>>(H1, mlp_w2, mlp_b2, H2, Nn, 512, 512, 1);
    router_kernel<<<(Nn + 7) / 8, blk, 0, stream>>>(H2, mlp_w3, mlp_b3, m, Nn);

    // result1, GCN layer 1
    result1_kernel<<<nb((size_t)Nn * 768), blk, 0, stream>>>(m, W1s, bp1, x2, result1, Nn);
    dim3 g256((256 + TILE_N - 1) / TILE_N, (Nn + TILE_M - 1) / TILE_M);
    gemm_bf16_wmma<<<g256, blk, 0, stream>>>(result1, gcn1_w, nullptr, h1, Nn, 256, 768, 0);
    zero_kernel<<<nb((size_t)Nn * 256), blk, 0, stream>>>(agg1, (size_t)Nn * 256);
    {   // F=256 -> shift=6, 4 edges per block
        unsigned blocks = (unsigned)((E + 3) / 4);
        scatter_kernel<<<blocks, blk, 0, stream>>>(h1, src, dst, dinv, agg1, E, 6);
    }
    gemm_bf16_wmma<<<g256, blk, 0, stream>>>(result1, W13, nullptr, r1w13, Nn, 256, 768, 0);
    g1_kernel<<<nb((size_t)Nn * 256), blk, 0, stream>>>(m, W12, gcn1_b, r1w13, g1, Nn);

    // result2, GCN layer 2
    result2_kernel<<<nb((size_t)Nn * 256), blk, 0, stream>>>(m, W2s, bp2, g1, result2, Nn);
    dim3 g32((32 + TILE_N - 1) / TILE_N, (Nn + TILE_M - 1) / TILE_M);
    gemm_bf16_wmma<<<g32, blk, 0, stream>>>(result2, gcn2_w, nullptr, h2, Nn, 32, 256, 0);
    zero_kernel<<<nb((size_t)Nn * 32), blk, 0, stream>>>(agg2, (size_t)Nn * 32);
    {   // F=32 -> shift=3, 32 edges per block
        unsigned blocks = (unsigned)((E + 31) / 32);
        scatter_kernel<<<blocks, blk, 0, stream>>>(h2, src, dst, dinv, agg2, E, 3);
    }

    // FC + log_softmax
    final_kernel<<<(Nn + 7) / 8, blk, 0, stream>>>(agg2, gcn2_b, fc_w, fc_b, (float*)d_out, Nn);
}
